// GroupedSlotAttention_46480136077838
// MI455X (gfx1250) — compile-verified, matrix operand order checked by reading, not run on hardware
//
#include <hip/hip_runtime.h>
#include <hip/hip_bf16.h>

typedef __attribute__((ext_vector_type(16))) __bf16 v16bf;
typedef __attribute__((ext_vector_type(8)))  float  v8f;

namespace {
constexpr int   B_   = 64;
constexpr int   F_   = 4096;
constexpr int   K_   = 8;
constexpr int   D_   = 192;
constexpr int   HID_ = 768;
constexpr float EPS_   = 1e-8f;
constexpr float SCALE_ = 0.07216878364870323f;  // 192^-0.5
constexpr float BIASF_ = -8.0f;                 // -SB_STRENGTH / (2 * 0.25^2)
constexpr int   LDA_  = 200;  // padded LDS ld for activation tile (conflict-free)
constexpr int   LDW_  = 200;  // padded LDS ld for weight panels
constexpr int   LDV_  = 136;  // padded LDS ld for v^T chunks
}

#define DEV static __device__ __forceinline__

DEV float wave_sum(float v) {
#pragma unroll
  for (int off = 16; off > 0; off >>= 1) v += __shfl_xor(v, off, 32);
  return v;
}

// 16x32 bf16 operand fragment from a row-major source with contiguous K.
// Used for A (rows = M) and for B given in B^T (rows = N) layout.
// Per-lane loads are two contiguous 16B chunks -> ds_load_b128/global_load_b128.
DEV v16bf frag_rowmajor(const __bf16* base, int ld, int c0, int lane) {
  const int r  = lane & 15;
  const int kb = (lane >> 4) << 3;  // 0 or 8
  const __bf16* p = base + (long)r * ld + c0 + kb;
  v16bf out;
#pragma unroll
  for (int i = 0; i < 8; ++i) out[i] = p[i];          // K = c0+kb .. +7
#pragma unroll
  for (int i = 0; i < 8; ++i) out[8 + i] = p[16 + i];  // K = c0+kb+16 .. +23
  return out;
}

// Async global->LDS 16B copy with portable fallback.
// Builtin signature (from hipcc diagnostic): param 1 is
// 'vector_size(16) int __device__ *' i.e. AS(1) pointer to GCC int4.
DEV void copy16_g2l(const __bf16* gsrc, __bf16* ldst) {
#if __has_builtin(__builtin_amdgcn_global_load_async_to_lds_b128)
  typedef int v4i_g __attribute__((vector_size(16)));
  typedef __attribute__((address_space(1))) v4i_g* gp_t;
  typedef __attribute__((address_space(3))) v4i_g* lp_t;
  __builtin_amdgcn_global_load_async_to_lds_b128((gp_t)gsrc, (lp_t)ldst, 0, 0);
#else
  *(uint4*)ldst = *(const uint4*)gsrc;
#endif
}

DEV void wait_async_copies() {
#if __has_builtin(__builtin_amdgcn_global_load_async_to_lds_b128)
#if __has_builtin(__builtin_amdgcn_s_wait_asynccnt)
  __builtin_amdgcn_s_wait_asynccnt(0);
#else
  asm volatile("s_wait_asynccnt 0" ::: "memory");
#endif
#endif
}

// ---------------------------------------------------------------------------
// prep: Wk/Wv -> bf16; transpose GRU weights for coalesced GEMV reads.
// ---------------------------------------------------------------------------
__global__ void prep_kernel(const float* __restrict__ Wk, const float* __restrict__ Wv,
                            const float* __restrict__ Wih, const float* __restrict__ Whh,
                            __bf16* __restrict__ wkb, __bf16* __restrict__ wvb,
                            float* __restrict__ wihT, float* __restrict__ whhT) {
  const int stride = gridDim.x * blockDim.x;
  for (int i = blockIdx.x * blockDim.x + threadIdx.x; i < D_ * D_; i += stride) {
    wkb[i] = (__bf16)Wk[i];
    wvb[i] = (__bf16)Wv[i];
  }
  for (int i = blockIdx.x * blockDim.x + threadIdx.x; i < 3 * 576 * D_; i += stride) {
    const int g = i / (576 * D_);
    const int rem = i - g * 576 * D_;
    const int n = rem / D_, d = rem - n * D_;
    wihT[g * 576 * D_ + d * 576 + n] = Wih[i];
    whhT[g * 576 * D_ + d * 576 + n] = Whh[i];
  }
}

// ---------------------------------------------------------------------------
// lnkv: LayerNorm(features) fused with k = f@Wk, v = f@Wv (bf16 WMMA).
// Activation tile + transposed weight panels in LDS; all fragment loads are
// contiguous 16B LDS reads on conflict-free padded strides.
// ---------------------------------------------------------------------------
__global__ __launch_bounds__(256) void lnkv_kernel(
    const float* __restrict__ feat, const float* __restrict__ lng,
    const float* __restrict__ lnb, const __bf16* __restrict__ wkb,
    const __bf16* __restrict__ wvb, __bf16* __restrict__ kb,
    __bf16* __restrict__ vb) {
  __shared__ __align__(16) __bf16 ldsA[128 * LDA_];      // 50.0 KB
  __shared__ __align__(16) __bf16 ldsWT[2 * 16 * LDW_];  // 12.5 KB (B^T panels)
  const int tid = threadIdx.x, lane = tid & 31, wave = tid >> 5;
  const long row0 = (long)blockIdx.x * 128;

  // LayerNorm 16 rows per wave, write normalized bf16 to LDS
  for (int rr = 0; rr < 16; ++rr) {
    const int rl = wave * 16 + rr;
    const float* fr = feat + (row0 + rl) * D_;
    float x[6];
    float s = 0.f;
#pragma unroll
    for (int j = 0; j < 6; ++j) { x[j] = fr[lane + 32 * j]; s += x[j]; }
    s = wave_sum(s);
    const float mean = s * (1.0f / D_);
    float vs = 0.f;
#pragma unroll
    for (int j = 0; j < 6; ++j) { const float d = x[j] - mean; vs += d * d; }
    vs = wave_sum(vs);
    const float inv = rsqrtf(vs * (1.0f / D_) + 1e-5f);
#pragma unroll
    for (int j = 0; j < 6; ++j) {
      const int d = lane + 32 * j;
      ldsA[rl * LDA_ + d] = (__bf16)((x[j] - mean) * inv * lng[d] + lnb[d]);
    }
  }
  __syncthreads();

  for (int nt = 0; nt < 12; ++nt) {
    // stage Wk / Wv column panels TRANSPOSED: ldsWT[mat][n][k]
    for (int i = tid; i < 2 * D_ * 16; i += 256) {
      const int mat = i / (D_ * 16);
      const int rem = i - mat * D_ * 16;
      const int d = rem >> 4, c = rem & 15;
      ldsWT[(mat * 16 + c) * LDW_ + d] = (mat ? wvb : wkb)[d * D_ + nt * 16 + c];
    }
    __syncthreads();
#pragma unroll
    for (int t = 0; t < 2; ++t) {
      const int tile = wave * 2 + t;
      const int mat = tile & 1, mtile = tile >> 1;
      v8f acc = {};
#pragma unroll
      for (int ks = 0; ks < 6; ++ks) {
        v16bf a = frag_rowmajor(ldsA + mtile * 16 * LDA_, LDA_, ks * 32, lane);
        v16bf b = frag_rowmajor(ldsWT + mat * 16 * LDW_, LDW_, ks * 32, lane);
        acc = __builtin_amdgcn_wmma_f32_16x16x32_bf16(false, a, false, b,
                                                      (short)0, acc, false, false);
      }
      __bf16* dst = mat ? vb : kb;
      const int rbase = mtile * 16 + ((lane >> 4) << 3);
      const int col = nt * 16 + (lane & 15);
#pragma unroll
      for (int r = 0; r < 8; ++r)
        dst[(row0 + rbase + r) * D_ + col] = (__bf16)acc[r];
    }
    __syncthreads();
  }
}

// ---------------------------------------------------------------------------
// transpose_v: v (B,F,D) -> vT (B,D,F), LDS-tiled 64x64, fully coalesced.
// ---------------------------------------------------------------------------
__global__ __launch_bounds__(256) void transpose_v_kernel(
    const __bf16* __restrict__ vb, __bf16* __restrict__ vT) {
  __shared__ __bf16 t[64][72];
  const int tx = threadIdx.x & 63, ty = threadIdx.x >> 6;  // 64 x 4
  const long b = blockIdx.z;
  const int f0 = blockIdx.x * 64, d0 = blockIdx.y * 64;
  const __bf16* src = vb + (b * F_ + f0) * D_ + d0;
#pragma unroll
  for (int i = 0; i < 16; ++i) {
    const int r = ty + i * 4;
    t[r][tx] = src[(long)r * D_ + tx];
  }
  __syncthreads();
  __bf16* dst = vT + (b * D_ + d0) * F_ + f0;
#pragma unroll
  for (int i = 0; i < 16; ++i) {
    const int r = ty + i * 4;
    dst[(long)r * F_ + tx] = t[tx][r];
  }
}

// ---------------------------------------------------------------------------
// qproj: LN(slots) @ Wq, writes zero-padded (B,16,D) bf16 q.
// ---------------------------------------------------------------------------
__global__ __launch_bounds__(192) void qproj_kernel(
    const float* __restrict__ slots, const float* __restrict__ lng,
    const float* __restrict__ lnb, const float* __restrict__ Wq,
    __bf16* __restrict__ qpad) {
  const int bk = blockIdx.x;  // 0..1023 = b*16 + kk
  const int b = bk >> 4, kk = bk & 15;
  const int t = threadIdx.x, lane = t & 31, wave = t >> 5;
  if (kk >= K_) { qpad[((long)b * 16 + kk) * D_ + t] = (__bf16)0.f; return; }
  __shared__ float sx[D_];
  __shared__ float red[8];
  const float x = slots[((long)b * K_ + kk) * D_ + t];
  float s = wave_sum(x);
  if (lane == 0) red[wave] = s;
  __syncthreads();
  if (t == 0) { float a = 0; for (int i = 0; i < 6; ++i) a += red[i]; red[6] = a; }
  __syncthreads();
  const float mean = red[6] * (1.0f / D_);
  const float dv = x - mean;
  float v2 = wave_sum(dv * dv);
  if (lane == 0) red[wave] = v2;
  __syncthreads();
  if (t == 0) { float a = 0; for (int i = 0; i < 6; ++i) a += red[i]; red[7] = a; }
  __syncthreads();
  const float inv = rsqrtf(red[7] * (1.0f / D_) + 1e-5f);
  sx[t] = dv * inv * lng[t] + lnb[t];
  __syncthreads();
  float acc = 0.f;
  for (int d = 0; d < D_; ++d) acc += sx[d] * Wq[d * D_ + t];
  qpad[((long)b * 16 + kk) * D_ + t] = (__bf16)acc;
}

// ---------------------------------------------------------------------------
// dots: per batch C(16x4096) = qpad(16x192) @ k^T, WMMA, fused spatial bias.
// Both operands read straight from global with contiguous per-lane 16B loads.
// ---------------------------------------------------------------------------
__global__ __launch_bounds__(256) void dots_kernel(
    const __bf16* __restrict__ qpad, const __bf16* __restrict__ kb,
    const float* __restrict__ centers, float* __restrict__ attn,
    int use_bias) {
  const int lane = threadIdx.x & 31, wave = threadIdx.x >> 5;
  const int b = blockIdx.y;
  const int f0 = (blockIdx.x * 8 + wave) * 16;
  const __bf16* qb = qpad + (long)b * 16 * D_;
  const __bf16* kbb = kb + ((long)b * F_ + f0) * D_;
  v8f acc = {};
#pragma unroll
  for (int ks = 0; ks < 6; ++ks) {
    v16bf a = frag_rowmajor(qb, D_, ks * 32, lane);
    v16bf bt = frag_rowmajor(kbb, D_, ks * 32, lane);  // B^T: rows = f, contiguous d
    acc = __builtin_amdgcn_wmma_f32_16x16x32_bf16(false, a, false, bt,
                                                  (short)0, acc, false, false);
  }
  if (lane < 16) {
    const int f = f0 + lane;
    const int xi = f & 63, yi = f >> 6;
    const float cx = -1.0f + 2.0f * xi / 63.0f;
    const float cy = -1.0f + 2.0f * yi / 63.0f;
#pragma unroll
    for (int r = 0; r < 8; ++r) {
      float val = acc[r] * SCALE_;
      if (use_bias) {
        const float mx = centers[(b * K_ + r) * 2 + 0];
        const float my = centers[(b * K_ + r) * 2 + 1];
        const float dx = cx - mx, dy = cy - my;
        val += BIASF_ * (dx * dx + dy * dy);
      }
      attn[((long)b * K_ + r) * F_ + f] = val;
    }
  }
}

__global__ void zero_rowsum_kernel(float* __restrict__ rowsum) {
  rowsum[threadIdx.x] = 0.f;
}

// ---------------------------------------------------------------------------
// softmax over the 8 slots per (b,f) + partial row sums via atomics.
// ---------------------------------------------------------------------------
__global__ __launch_bounds__(256) void softmax_kernel(
    float* __restrict__ attn, float* __restrict__ rowsum) {
  const int tid = threadIdx.x, lane = tid & 31, wave = tid >> 5;
  const int f = blockIdx.x * 256 + tid;
  const int b = blockIdx.y;
  __shared__ float red[8];
  float v[K_];
  float mx = -1e30f;
#pragma unroll
  for (int k = 0; k < K_; ++k) {
    v[k] = attn[((long)b * K_ + k) * F_ + f];
    mx = fmaxf(mx, v[k]);
  }
  float s = 0.f;
#pragma unroll
  for (int k = 0; k < K_; ++k) { v[k] = __expf(v[k] - mx); s += v[k]; }
  const float inv = 1.0f / s;
#pragma unroll
  for (int k = 0; k < K_; ++k) {
    v[k] *= inv;
    attn[((long)b * K_ + k) * F_ + f] = v[k];
  }
  for (int k = 0; k < K_; ++k) {
    const float ps = wave_sum(v[k]);
    if (lane == 0) red[wave] = ps;
    __syncthreads();
    if (tid == 0) {
      float t = 0;
      for (int i = 0; i < 8; ++i) t += red[i];
      atomicAdd(&rowsum[b * K_ + k], t);
    }
    __syncthreads();
  }
}

// ---------------------------------------------------------------------------
// normw: w = attn / (rowsum + eps) as bf16, rows 8..15 zero-padded.
// ---------------------------------------------------------------------------
__global__ __launch_bounds__(256) void normw_kernel(
    const float* __restrict__ attn, const float* __restrict__ rowsum,
    __bf16* __restrict__ wpad) {
  const int f = blockIdx.x * 256 + threadIdx.x;
  const int bkk = blockIdx.y;
  const int b = bkk >> 4, kk = bkk & 15;
  __bf16 w = (__bf16)0.f;
  if (kk < K_) {
    const float s = rowsum[b * K_ + kk] + EPS_;
    w = (__bf16)(attn[((long)b * K_ + kk) * F_ + f] / s);
  }
  wpad[((long)b * 16 + kk) * F_ + f] = w;
}

// ---------------------------------------------------------------------------
// centers: soft-assignment centroids for next iteration's spatial bias.
// ---------------------------------------------------------------------------
__global__ __launch_bounds__(256) void centers_kernel(
    const float* __restrict__ attn, const float* __restrict__ rowsum,
    float* __restrict__ centers) {
  const int b = blockIdx.x >> 3, kk = blockIdx.x & 7;
  const int tid = threadIdx.x, lane = tid & 31, wave = tid >> 5;
  const float* ar = attn + ((long)b * K_ + kk) * F_;
  float cx = 0.f, cy = 0.f;
  for (int f = tid; f < F_; f += 256) {
    const float a = ar[f];
    const int xi = f & 63, yi = f >> 6;
    cx += a * (-1.0f + 2.0f * xi / 63.0f);
    cy += a * (-1.0f + 2.0f * yi / 63.0f);
  }
  cx = wave_sum(cx);
  cy = wave_sum(cy);
  __shared__ float rx[8], ry[8];
  if (lane == 0) { rx[wave] = cx; ry[wave] = cy; }
  __syncthreads();
  if (tid == 0) {
    float sx = 0.f, sy = 0.f;
    for (int i = 0; i < 8; ++i) { sx += rx[i]; sy += ry[i]; }
    const float inv = 1.0f / (rowsum[b * K_ + kk] + EPS_);
    centers[(b * K_ + kk) * 2 + 0] = sx * inv;
    centers[(b * K_ + kk) * 2 + 1] = sy * inv;
  }
}

// ---------------------------------------------------------------------------
// updates: per batch C(16x192) = wpad(16x4096) @ v(4096x192), WMMA.
// v^T chunks (192 x 128, padded) staged into LDS with async 16B copies when
// available; all fragment reads are contiguous 16B on conflict-free strides.
// ---------------------------------------------------------------------------
__global__ __launch_bounds__(384) void updates_kernel(
    const __bf16* __restrict__ wpad, const __bf16* __restrict__ vT,
    float* __restrict__ upd) {
  __shared__ __align__(16) __bf16 vt[D_ * LDV_];  // 51 KB
  const int tid = threadIdx.x, lane = tid & 31, wave = tid >> 5;
  const int b = blockIdx.x;
  const int n0 = wave * 16;
  const __bf16* wrow = wpad + (long)b * 16 * F_;
  v8f acc = {};
  for (int chunk = 0; chunk < 32; ++chunk) {
    // stage vT[b][0..191][chunk*128 .. +128) -> vt[d][fo], 16B granules
    for (int i = tid; i < D_ * 16; i += 384) {   // 192 rows x 16 chunks of 8
      const int d = i >> 4;
      const int fo = (i & 15) * 8;
      copy16_g2l(vT + ((long)b * D_ + d) * F_ + chunk * 128 + fo,
                 vt + d * LDV_ + fo);
    }
    wait_async_copies();
    __syncthreads();
#pragma unroll
    for (int ks = 0; ks < 4; ++ks) {
      v16bf a = frag_rowmajor(wrow, F_, chunk * 128 + ks * 32, lane);
      v16bf bt = frag_rowmajor(vt + n0 * LDV_, LDV_, ks * 32, lane);  // rows = n(d)
      acc = __builtin_amdgcn_wmma_f32_16x16x32_bf16(false, a, false, bt,
                                                    (short)0, acc, false, false);
    }
    __syncthreads();
  }
  if (lane < 16) {
#pragma unroll
    for (int r = 0; r < 8; ++r)
      upd[((long)b * K_ + r) * D_ + n0 + lane] = acc[r];
  }
}

// ---------------------------------------------------------------------------
// slot_update: fused GRU cell + LayerNorm + MLP residual per (b, slot).
// ---------------------------------------------------------------------------
__global__ __launch_bounds__(192) void slot_update_kernel(
    const float* __restrict__ upd, const float* __restrict__ slots_prev,
    const float* __restrict__ wihT, const float* __restrict__ whhT,
    const float* __restrict__ bih, const float* __restrict__ bhh,
    const float* __restrict__ mlng, const float* __restrict__ mlnb,
    const float* __restrict__ W1, const float* __restrict__ b1,
    const float* __restrict__ W2, const float* __restrict__ b2,
    float* __restrict__ slots_out) {
  const int bk = blockIdx.x, kk = bk & 7;
  const int g = (kk == 0) ? 0 : (kk < 7 ? 1 : 2);
  const int t = threadIdx.x, lane = t & 31, wave = t >> 5;
  __shared__ float sx[D_], sh[D_], shid[HID_], red[8];
  const float h_prev = slots_prev[(long)bk * D_ + t];
  sx[t] = upd[(long)bk * D_ + t];
  sh[t] = h_prev;
  __syncthreads();
  const float* wih = wihT + g * D_ * 576;
  const float* whh = whhT + g * D_ * 576;
  float gi_r = bih[g * 576 + t], gi_z = bih[g * 576 + 192 + t], gi_n = bih[g * 576 + 384 + t];
  float gh_r = bhh[g * 576 + t], gh_z = bhh[g * 576 + 192 + t], gh_n = bhh[g * 576 + 384 + t];
  for (int d = 0; d < D_; ++d) {
    const float xv = sx[d], hv = sh[d];
    const float* wi = wih + d * 576;
    const float* wh = whh + d * 576;
    gi_r += xv * wi[t];        gh_r += hv * wh[t];
    gi_z += xv * wi[192 + t];  gh_z += hv * wh[192 + t];
    gi_n += xv * wi[384 + t];  gh_n += hv * wh[384 + t];
  }
  const float r = 1.0f / (1.0f + __expf(-(gi_r + gh_r)));
  const float z = 1.0f / (1.0f + __expf(-(gi_z + gh_z)));
  const float nn = tanhf(gi_n + r * gh_n);
  const float h = (1.0f - z) * nn + z * h_prev;
  // LayerNorm(h)
  float s = wave_sum(h);
  if (lane == 0) red[wave] = s;
  __syncthreads();
  if (t == 0) { float a = 0; for (int i = 0; i < 6; ++i) a += red[i]; red[6] = a; }
  __syncthreads();
  const float mean = red[6] * (1.0f / D_);
  const float dv = h - mean;
  float v2 = wave_sum(dv * dv);
  if (lane == 0) red[wave] = v2;
  __syncthreads();
  if (t == 0) { float a = 0; for (int i = 0; i < 6; ++i) a += red[i]; red[7] = a; }
  __syncthreads();
  const float inv = rsqrtf(red[7] * (1.0f / D_) + 1e-5f);
  sx[t] = dv * inv * mlng[g * D_ + t] + mlnb[g * D_ + t];
  __syncthreads();
  const float* w1 = W1 + g * D_ * HID_;
#pragma unroll
  for (int p = 0; p < 4; ++p) {
    const int j = t + 192 * p;
    float a = b1[g * HID_ + j];
    for (int d = 0; d < D_; ++d) a += sx[d] * w1[d * HID_ + j];
    shid[j] = fmaxf(a, 0.0f);
  }
  __syncthreads();
  const float* w2 = W2 + g * HID_ * D_;
  float o = b2[g * D_ + t];
  for (int d = 0; d < HID_; ++d) o += shid[d] * w2[d * D_ + t];
  slots_out[(long)bk * D_ + t] = h + o;
}

// ---------------------------------------------------------------------------
extern "C" void kernel_launch(void* const* d_in, const int* in_sizes, int n_in,
                              void* d_out, int out_size, void* d_ws, size_t ws_size,
                              hipStream_t stream) {
  (void)in_sizes; (void)n_in; (void)out_size; (void)ws_size;
  const float* features = (const float*)d_in[0];
  const float* slots_in = (const float*)d_in[1];
  const float* Wk   = (const float*)d_in[2];
  const float* Wv   = (const float*)d_in[3];
  const float* Wq   = (const float*)d_in[4];
  const float* lnfg = (const float*)d_in[5];
  const float* lnfb = (const float*)d_in[6];
  const float* lnsg = (const float*)d_in[7];
  const float* lnsb = (const float*)d_in[8];
  const float* gWih = (const float*)d_in[9];
  const float* gWhh = (const float*)d_in[10];
  const float* gbih = (const float*)d_in[11];
  const float* gbhh = (const float*)d_in[12];
  const float* mlng = (const float*)d_in[13];
  const float* mlnb = (const float*)d_in[14];
  const float* W1   = (const float*)d_in[15];
  const float* b1   = (const float*)d_in[16];
  const float* W2   = (const float*)d_in[17];
  const float* b2   = (const float*)d_in[18];

  char* p = (char*)d_ws;
  auto alloc = [&](size_t bytes) -> void* {
    void* r = (void*)p;
    p += (bytes + 255) & ~size_t(255);
    return r;
  };
  __bf16* kb     = (__bf16*)alloc((size_t)B_ * F_ * D_ * 2);
  __bf16* vb     = (__bf16*)alloc((size_t)B_ * F_ * D_ * 2);
  __bf16* vTb    = (__bf16*)alloc((size_t)B_ * F_ * D_ * 2);
  float*  attn   = (float*) alloc((size_t)B_ * K_ * F_ * 4);
  __bf16* wpad   = (__bf16*)alloc((size_t)B_ * 16 * F_ * 2);
  __bf16* qpad   = (__bf16*)alloc((size_t)B_ * 16 * D_ * 2);
  float*  upd    = (float*) alloc((size_t)B_ * K_ * D_ * 4);
  float*  slotsA = (float*) alloc((size_t)B_ * K_ * D_ * 4);
  float*  slotsB = (float*) alloc((size_t)B_ * K_ * D_ * 4);
  float*  centers= (float*) alloc((size_t)B_ * K_ * 2 * 4);
  float*  rowsum = (float*) alloc((size_t)B_ * K_ * 4);
  __bf16* wkb    = (__bf16*)alloc((size_t)D_ * D_ * 2);
  __bf16* wvb    = (__bf16*)alloc((size_t)D_ * D_ * 2);
  float*  wihT   = (float*) alloc((size_t)3 * 576 * D_ * 4);
  float*  whhT   = (float*) alloc((size_t)3 * 576 * D_ * 4);

  prep_kernel<<<256, 256, 0, stream>>>(Wk, Wv, gWih, gWhh, wkb, wvb, wihT, whhT);
  lnkv_kernel<<<B_ * F_ / 128, 256, 0, stream>>>(features, lnfg, lnfb, wkb, wvb, kb, vb);
  transpose_v_kernel<<<dim3(F_ / 64, D_ / 64, B_), 256, 0, stream>>>(vb, vTb);

  for (int it = 0; it < 3; ++it) {
    const float* sl_prev = (it == 0) ? slots_in : (it == 1 ? slotsA : slotsB);
    float* sl_out = (it == 0) ? slotsA : (it == 1 ? slotsB : (float*)d_out);

    qproj_kernel<<<B_ * 16, 192, 0, stream>>>(sl_prev, lnsg, lnsb, Wq, qpad);
    dots_kernel<<<dim3(32, B_), 256, 0, stream>>>(qpad, kb, centers, attn, it > 0 ? 1 : 0);
    zero_rowsum_kernel<<<1, B_ * K_, 0, stream>>>(rowsum);
    softmax_kernel<<<dim3(F_ / 256, B_), 256, 0, stream>>>(attn, rowsum);
    normw_kernel<<<dim3(F_ / 256, B_ * 16), 256, 0, stream>>>(attn, rowsum, wpad);
    centers_kernel<<<B_ * K_, 256, 0, stream>>>(attn, rowsum, centers);
    updates_kernel<<<B_, 384, 0, stream>>>(wpad, vTb, upd);
    slot_update_kernel<<<B_ * K_, 192, 0, stream>>>(
        upd, sl_prev, wihT, whhT, gbih, gbhh, mlng, mlnb, W1, b1, W2, b2, sl_out);
  }
}
